// HierarchicalDecoder_9835475108281
// MI455X (gfx1250) — matrix operand order repaired and checked by reading, not compiled, not run on hardware
//
#include <hip/hip_runtime.h>
#include <hip/hip_bf16.h>

// ---------------------------------------------------------------------------
// Problem constants (from reference)
// ---------------------------------------------------------------------------
#define BATCH 2048
#define ZD    256
#define HDIM  1024
#define H3    3072
#define EDIM  96
#define VDIM  512
#define XK    1120   // E + H
#define NBEAT 4
#define NTICK 6
#define NSTEP 24

typedef __bf16 bf16_t;
typedef __attribute__((ext_vector_type(16))) __bf16 v16bf;
typedef __attribute__((ext_vector_type(8)))  __bf16 v8bf;
typedef __attribute__((ext_vector_type(8)))  float  v8f;

#define ACT_NONE 0
#define ACT_SELU 1
#define ACT_RELU 2

// ---------------------------------------------------------------------------
// bf16 WMMA GEMM:  out[M,N] = act( A[M,K] * Wt[N,K]^T + bias[N] )
//
// Block: 256 threads (8 waves, 2x4). Block tile 64(M) x 256(N), K-step 32.
// Each wave computes a 32x64 patch as 2x4 v_wmma_f32_16x16x32_bf16 tiles.
// Staging uses CDNA5 async copies (global_load_async_to_lds_b128, ASYNCcnt)
// with double-buffered LDS: tile k+1 streams into LDS while WMMAs run on k.
// ---------------------------------------------------------------------------
#define TILE_M 64
#define TILE_N 256
#define TILE_K 32
#define LDSK   40     // padded row stride in halves (80B, multiple of 16B)

__device__ __forceinline__ void async_cp16(const bf16_t* gsrc, const bf16_t* ldst)
{
    // GLOBAL_LOAD_ASYNC_TO_LDS_B128: vdst = LDS byte address, vaddr = 64-bit
    // global address (GV mode). Tracked with ASYNCcnt.
    unsigned lds_off = (unsigned)(unsigned long long)(const void*)ldst;
    unsigned long long ga = (unsigned long long)(const void*)gsrc;
    asm volatile("global_load_async_to_lds_b128 %0, %1, off"
                 :: "v"(lds_off), "v"(ga) : "memory");
}

__device__ __forceinline__ void wait_async0()
{
    asm volatile("s_wait_asynccnt 0" ::: "memory");
}

template <int ACT>
__launch_bounds__(256, 2)
__global__ void gemm_bf16_act(const bf16_t* __restrict__ A, int lda,
                              const bf16_t* __restrict__ Wt, int ldw,
                              const float* __restrict__ bias,
                              float* __restrict__ outF, long ldoF,
                              bf16_t* __restrict__ outB, long ldoB,
                              int M, int N, int K)
{
    (void)M; (void)N;
    __shared__ __align__(16) bf16_t sA[2][TILE_M][LDSK];
    __shared__ __align__(16) bf16_t sB[2][TILE_N][LDSK];   // N-major: sB[..][n][k]

    const int tid   = threadIdx.x;
    const int lane  = tid & 31;
    const int wave  = tid >> 5;
    const int wm    = wave >> 2;          // 0..1 (M sub-tile of wave)
    const int wn    = wave & 3;           // 0..3 (N sub-tile of wave)
    const int blkN  = blockIdx.x * TILE_N;
    const int blkM  = blockIdx.y * TILE_M;

    const int row16 = lane & 15;
    const int hi    = lane >> 4;          // 0 or 1 (lane half)

    // accumulators preloaded with bias (C/D layout: lane -> N column)
    v8f acc[2][4];
    float bv[4];
#pragma unroll
    for (int ni = 0; ni < 4; ++ni)
        bv[ni] = bias ? bias[blkN + wn * 64 + ni * 16 + row16] : 0.0f;
#pragma unroll
    for (int mi = 0; mi < 2; ++mi)
#pragma unroll
        for (int ni = 0; ni < 4; ++ni)
#pragma unroll
            for (int v = 0; v < 8; ++v) acc[mi][ni][v] = bv[ni];

    // staging assignments (each thread: 1x16B of A, 4x16B of B per K-step)
    const int arow = tid >> 2;            // 0..63
    const int aseg = (tid & 3) * 8;       // 0,8,16,24 halves
    const int brow = tid;                 // 0..255

    const bf16_t* gA = A + (long)(blkM + arow) * lda + aseg;
    const bf16_t* gB = Wt + (long)(blkN + brow) * ldw;

    auto stage = [&](int buf, int k0) {
        async_cp16(gA + k0, &sA[buf][arow][aseg]);
#pragma unroll
        for (int s = 0; s < 4; ++s)
            async_cp16(gB + k0 + s * 8, &sB[buf][brow][s * 8]);
    };

    stage(0, 0);
    wait_async0();
    __syncthreads();

    const int nk = K / TILE_K;
    for (int it = 0; it < nk; ++it) {
        const int cur = it & 1;
        if (it + 1 < nk) stage(cur ^ 1, (it + 1) * TILE_K);

        // A fragment: lane half 0 -> K 0..7 & 16..23; half 1 -> K 8..15 & 24..31
        v16bf afrag[2], bfrag[4];
        const int akb = hi ? 8 : 0;
#pragma unroll
        for (int mi = 0; mi < 2; ++mi) {
            const bf16_t* ap = &sA[cur][wm * 32 + mi * 16 + row16][0];
            const v8bf a0 = *reinterpret_cast<const v8bf*>(ap + akb);
            const v8bf a1 = *reinterpret_cast<const v8bf*>(ap + akb + 16);
#pragma unroll
            for (int j = 0; j < 8; ++j) { afrag[mi][j] = a0[j]; afrag[mi][8 + j] = a1[j]; }
        }
        // B fragment: lane half 0 -> K 0..7 & 16..23; half 1 -> K 8..15 & 24..31 (column = lane%16)
        const int bkb = hi ? 8 : 0;
#pragma unroll
        for (int ni = 0; ni < 4; ++ni) {
            const bf16_t* bp = &sB[cur][wn * 64 + ni * 16 + row16][0];
            const v8bf b0 = *reinterpret_cast<const v8bf*>(bp + bkb);
            const v8bf b1 = *reinterpret_cast<const v8bf*>(bp + bkb + 16);
#pragma unroll
            for (int j = 0; j < 8; ++j) { bfrag[ni][j] = b0[j]; bfrag[ni][8 + j] = b1[j]; }
        }

#pragma unroll
        for (int mi = 0; mi < 2; ++mi)
#pragma unroll
            for (int ni = 0; ni < 4; ++ni)
                acc[mi][ni] = __builtin_amdgcn_wmma_f32_16x16x32_bf16(
                    false, afrag[mi], false, bfrag[ni],
                    (short)0, acc[mi][ni], false, false);

        wait_async0();      // next-tile async copies fully landed in LDS
        __syncthreads();    // all waves done reading cur / writing nxt
    }

    // epilogue: activation + stores (f32 and/or bf16)
#pragma unroll
    for (int mi = 0; mi < 2; ++mi) {
#pragma unroll
        for (int ni = 0; ni < 4; ++ni) {
            const int n = blkN + wn * 64 + ni * 16 + row16;
#pragma unroll
            for (int v = 0; v < 8; ++v) {
                const int m = blkM + wm * 32 + mi * 16 + hi * 8 + v;
                float x = acc[mi][ni][v];
                if (ACT == ACT_SELU) {
                    const float scale = 1.0507009873554805f;
                    const float alpha = 1.6732632423543772f;
                    x = scale * (x > 0.f ? x : alpha * (__expf(x) - 1.f));
                } else if (ACT == ACT_RELU) {
                    x = x > 0.f ? x : 0.f;
                }
                if (outF) outF[(long)m * ldoF + n] = x;
                if (outB) outB[(long)m * ldoB + n] = (bf16_t)x;
            }
        }
    }
}

// ---------------------------------------------------------------------------
// Elementwise GRU gate fusion
// ---------------------------------------------------------------------------
__device__ __forceinline__ float sigmoidf_(float x) { return 1.f / (1.f + __expf(-x)); }

// beat GRU: input path is rank-1 (b_in = broadcast(b0)), gi = b0*Wi + bi
__global__ void beat_combine(const float* __restrict__ Wi, const float* __restrict__ bi,
                             const float* __restrict__ b0,
                             const float* __restrict__ gh,   // B x 3H (bh added by GEMM)
                             float* __restrict__ h,          // B x H  in/out
                             bf16_t* __restrict__ hout)      // B x H  bf16
{
    long idx = (long)blockIdx.x * blockDim.x + threadIdx.x;
    if (idx >= (long)BATCH * HDIM) return;
    int b = (int)(idx / HDIM), n = (int)(idx % HDIM);
    float b0v = b0[0];
    float ir = b0v * Wi[n]            + bi[n];
    float iz = b0v * Wi[HDIM + n]     + bi[HDIM + n];
    float in = b0v * Wi[2 * HDIM + n] + bi[2 * HDIM + n];
    const float* ghr = gh + (long)b * H3;
    float r  = sigmoidf_(ir + ghr[n]);
    float zg = sigmoidf_(iz + ghr[HDIM + n]);
    float nn = tanhf(in + r * ghr[2 * HDIM + n]);
    float o  = (1.f - zg) * nn + zg * h[idx];
    h[idx] = o;
    hout[idx] = (bf16_t)o;
}

// tick GRU: both gi and gh are full B x 3H matrices (biases already added)
__global__ void tick_combine(const float* __restrict__ gi, const float* __restrict__ gh,
                             float* __restrict__ h, bf16_t* __restrict__ hout)
{
    long idx = (long)blockIdx.x * blockDim.x + threadIdx.x;
    if (idx >= (long)BATCH * HDIM) return;
    int b = (int)(idx / HDIM), n = (int)(idx % HDIM);
    const float* gir = gi + (long)b * H3;
    const float* ghr = gh + (long)b * H3;
    float r  = sigmoidf_(gir[n] + ghr[n]);
    float zg = sigmoidf_(gir[HDIM + n] + ghr[HDIM + n]);
    float nn = tanhf(gir[2 * HDIM + n] + r * ghr[2 * HDIM + n]);
    float o  = (1.f - zg) * nn + zg * h[idx];
    h[idx] = o;
    hout[idx] = (bf16_t)o;
}

// ---------------------------------------------------------------------------
// argmax over V logits (first-max tie-break, like jnp.argmax) + emb gather
// writes int64 sample and bf16 embedding into xcat[:, 0:E]
// ---------------------------------------------------------------------------
__global__ void argmax_gather(const float* __restrict__ logits, long ldl,
                              const float* __restrict__ emb,
                              float* __restrict__ samples_base, int step,
                              bf16_t* __restrict__ xcat)
{
    __shared__ float svals[256];
    __shared__ int   sidx[256];
    __shared__ int   widx;
    const int b = blockIdx.x;
    const int t = threadIdx.x;
    const float* row = logits + (long)b * ldl;
    float best = -3.4e38f; int bidx = 0;
    for (int v = t; v < VDIM; v += 256) {
        float x = row[v];
        if (x > best) { best = x; bidx = v; }
    }
    svals[t] = best; sidx[t] = bidx;
    __syncthreads();
    for (int s = 128; s > 0; s >>= 1) {
        if (t < s) {
            float a = svals[t], c = svals[t + s];
            if (c > a || (c == a && sidx[t + s] < sidx[t])) {
                svals[t] = c; sidx[t] = sidx[t + s];
            }
        }
        __syncthreads();
    }
    if (t == 0) {
        widx = sidx[0];
        ((long long*)samples_base)[(long)b * NSTEP + step] = (long long)sidx[0];
    }
    __syncthreads();
    const int id = widx;
    for (int e = t; e < EDIM; e += 256)
        xcat[(long)b * XK + e] = (bf16_t)emb[(long)id * EDIM + e];
}

// ---------------------------------------------------------------------------
// weight / activation casts
// ---------------------------------------------------------------------------
__global__ void cast_direct(const float* __restrict__ src, bf16_t* __restrict__ dst, long n)
{
    long i = (long)blockIdx.x * blockDim.x + threadIdx.x;
    if (i < n) dst[i] = (bf16_t)src[i];
}

// src is (K,N) row-major f32; dst is (N,K) row-major bf16
__global__ void cast_transpose(const float* __restrict__ src, bf16_t* __restrict__ dst,
                               int K, int N)
{
    long i = (long)blockIdx.x * blockDim.x + threadIdx.x;
    if (i >= (long)K * N) return;
    int k = (int)(i / N), n = (int)(i % N);
    dst[(long)n * K + k] = (bf16_t)src[i];
}

__global__ void init_xcat(const float* __restrict__ x0, bf16_t* __restrict__ xcat)
{
    long i = (long)blockIdx.x * blockDim.x + threadIdx.x;
    if (i >= (long)BATCH * EDIM) return;
    int b = (int)(i / EDIM), e = (int)(i % EDIM);
    xcat[(long)b * XK + e] = (bf16_t)x0[e];
}

// ---------------------------------------------------------------------------
// host orchestration
// ---------------------------------------------------------------------------
extern "C" void kernel_launch(void* const* d_in, const int* in_sizes, int n_in,
                              void* d_out, int out_size, void* d_ws, size_t ws_size,
                              hipStream_t stream)
{
    (void)in_sizes; (void)n_in; (void)out_size; (void)ws_size;

    const float* z    = (const float*)d_in[0];
    // d_in[1] score_tensor, d_in[2] train: unused at inference (train=0)
    const float* emb  = (const float*)d_in[3];
    const float* z2W  = (const float*)d_in[4];   // (ZD, H)
    const float* z2b  = (const float*)d_in[5];
    const float* b0   = (const float*)d_in[6];
    const float* bWi  = (const float*)d_in[7];   // (3H, 1)
    const float* bWh  = (const float*)d_in[8];   // (3H, H)  already N-major
    const float* bbi  = (const float*)d_in[9];
    const float* bbh  = (const float*)d_in[10];
    const float* hidW = (const float*)d_in[11];  // (H, H)
    const float* hidb = (const float*)d_in[12];
    const float* inW  = (const float*)d_in[13];  // (H, H)
    const float* inb  = (const float*)d_in[14];
    const float* x0   = (const float*)d_in[15];
    const float* tWi  = (const float*)d_in[16];  // (3H, XK) already N-major
    const float* tWh  = (const float*)d_in[17];  // (3H, H)  already N-major
    const float* tbi  = (const float*)d_in[18];
    const float* tbh  = (const float*)d_in[19];
    const float* oW   = (const float*)d_in[20];  // (H, V)
    const float* ob   = (const float*)d_in[21];

    // workspace carve-up (256B aligned)
    char* wsb = (char*)d_ws;
    size_t off = 0;
    auto carve = [&](size_t bytes) -> char* {
        size_t o = (off + 255) & ~(size_t)255;
        off = o + bytes;
        return wsb + o;
    };
    bf16_t* zbf    = (bf16_t*)carve((size_t)BATCH * ZD * 2);
    bf16_t* z2bT   = (bf16_t*)carve((size_t)HDIM * ZD * 2);
    bf16_t* bWhT   = (bf16_t*)carve((size_t)H3 * HDIM * 2);
    bf16_t* hidT   = (bf16_t*)carve((size_t)HDIM * HDIM * 2);
    bf16_t* inT    = (bf16_t*)carve((size_t)HDIM * HDIM * 2);
    bf16_t* tWiT   = (bf16_t*)carve((size_t)H3 * XK * 2);
    bf16_t* tWhT   = (bf16_t*)carve((size_t)H3 * HDIM * 2);
    bf16_t* oWT    = (bf16_t*)carve((size_t)VDIM * HDIM * 2);
    float*  hF     = (float*) carve((size_t)BATCH * HDIM * 4);
    bf16_t* hBF    = (bf16_t*)carve((size_t)BATCH * HDIM * 2);
    bf16_t* beatBF = (bf16_t*)carve((size_t)NBEAT * BATCH * HDIM * 2);
    float*  ghF    = (float*) carve((size_t)BATCH * H3 * 4);
    float*  giF    = (float*) carve((size_t)BATCH * H3 * 4);
    bf16_t* xcat   = (bf16_t*)carve((size_t)BATCH * XK * 2);

    auto blk = [](long n) { return (unsigned)((n + 255) / 256); };

    // ---- weight prep (bf16, N-major) ----
    cast_direct<<<blk((long)BATCH * ZD), 256, 0, stream>>>(z, zbf, (long)BATCH * ZD);
    cast_transpose<<<blk((long)ZD * HDIM), 256, 0, stream>>>(z2W, z2bT, ZD, HDIM);
    cast_direct<<<blk((long)H3 * HDIM), 256, 0, stream>>>(bWh, bWhT, (long)H3 * HDIM);
    cast_transpose<<<blk((long)HDIM * HDIM), 256, 0, stream>>>(hidW, hidT, HDIM, HDIM);
    cast_transpose<<<blk((long)HDIM * HDIM), 256, 0, stream>>>(inW, inT, HDIM, HDIM);
    cast_direct<<<blk((long)H3 * XK), 256, 0, stream>>>(tWi, tWiT, (long)H3 * XK);
    cast_direct<<<blk((long)H3 * HDIM), 256, 0, stream>>>(tWh, tWhT, (long)H3 * HDIM);
    cast_transpose<<<blk((long)HDIM * VDIM), 256, 0, stream>>>(oW, oWT, HDIM, VDIM);
    init_xcat<<<blk((long)BATCH * EDIM), 256, 0, stream>>>(x0, xcat);

    auto gemm = [&](const bf16_t* A, int lda, const bf16_t* Wt, int ldw,
                    const float* bias, float* outF, long ldoF,
                    bf16_t* outB, long ldoB, int M, int N, int K, int act) {
        dim3 grid((unsigned)(N / TILE_N), (unsigned)(M / TILE_M));
        if (act == ACT_SELU)
            gemm_bf16_act<ACT_SELU><<<grid, 256, 0, stream>>>(A, lda, Wt, ldw, bias,
                outF, ldoF, outB, ldoB, M, N, K);
        else if (act == ACT_RELU)
            gemm_bf16_act<ACT_RELU><<<grid, 256, 0, stream>>>(A, lda, Wt, ldw, bias,
                outF, ldoF, outB, ldoB, M, N, K);
        else
            gemm_bf16_act<ACT_NONE><<<grid, 256, 0, stream>>>(A, lda, Wt, ldw, bias,
                outF, ldoF, outB, ldoB, M, N, K);
    };

    // ---- h_beat = selu(z @ z2beat_W + b) ----
    gemm(zbf, ZD, z2bT, ZD, z2b, hF, HDIM, hBF, HDIM, BATCH, HDIM, ZD, ACT_SELU);

    // ---- beat GRU (4 sequential steps) ----
    for (int i = 0; i < NBEAT; ++i) {
        const bf16_t* hin = (i == 0) ? hBF : beatBF + (size_t)(i - 1) * BATCH * HDIM;
        gemm(hin, HDIM, bWhT, HDIM, bbh, ghF, H3, nullptr, 0, BATCH, H3, HDIM, ACT_NONE);
        beat_combine<<<blk((long)BATCH * HDIM), 256, 0, stream>>>(
            bWi, bbi, b0, ghF, hF, beatBF + (size_t)i * BATCH * HDIM);
    }

    float* outW    = (float*)d_out;                       // (B, 24, V) f32
    float* samples = outW + (size_t)BATCH * NSTEP * VDIM; // (B, 1, 24) int64

    // ---- per-beat tick decoding ----
    for (int i = 0; i < NBEAT; ++i) {
        const bf16_t* ctx = beatBF + (size_t)i * BATCH * HDIM;
        // h = selu(ctx @ b2t_hid_W + b)
        gemm(ctx, HDIM, hidT, HDIM, hidb, hF, HDIM, hBF, HDIM, BATCH, HDIM, HDIM, ACT_SELU);
        // beat_emb = selu(ctx @ b2t_in_W + b) -> bf16 straight into xcat[:, E:]
        gemm(ctx, HDIM, inT, HDIM, inb, nullptr, 0, xcat + EDIM, XK, BATCH, HDIM, HDIM, ACT_SELU);

        for (int t = 0; t < NTICK; ++t) {
            const int step = i * NTICK + t;
            // gi = [x_emb | beat_emb] @ tick_Wi.T + bi  (single K=1120 GEMM)
            gemm(xcat, XK, tWiT, XK, tbi, giF, H3, nullptr, 0, BATCH, H3, XK, ACT_NONE);
            // gh = h @ tick_Wh.T + bh
            gemm(hBF, HDIM, tWhT, HDIM, tbh, ghF, H3, nullptr, 0, BATCH, H3, HDIM, ACT_NONE);
            tick_combine<<<blk((long)BATCH * HDIM), 256, 0, stream>>>(giF, ghF, hF, hBF);
            // logits = relu(h @ out_W + b) directly into d_out at (b, step, :)
            gemm(hBF, HDIM, oWT, HDIM, ob,
                 outW + (size_t)step * VDIM, (long)NSTEP * VDIM,
                 nullptr, 0, BATCH, VDIM, HDIM, ACT_RELU);
            // argmax + embedding gather -> xcat[:, 0:E], samples int64
            argmax_gather<<<BATCH, 256, 0, stream>>>(
                outW + (size_t)step * VDIM, (long)NSTEP * VDIM, emb, samples, step, xcat);
        }
    }
}
